// HadamardQuantLinear_50319836840176
// MI455X (gfx1250) — 8-bit matrix operand order repaired and checked by reading, compiled, not run on hardware
//
#include <hip/hip_runtime.h>
#include <hip/hip_bf16.h>
#include <stdint.h>

typedef __attribute__((ext_vector_type(16))) int   v16i;
typedef __attribute__((ext_vector_type(8)))  float v8f;

#if __has_builtin(__builtin_amdgcn_global_load_async_to_lds_b128)
#define USE_ASYNC_LDS 1
typedef int v4ib __attribute__((vector_size(16)));
typedef __attribute__((address_space(1))) v4ib GV4;
typedef __attribute__((address_space(3))) v4ib LV4;
#else
#define USE_ASYNC_LDS 0
#endif

template <int N>
__device__ __forceinline__ void wait_asynccnt() {
#if __has_builtin(__builtin_amdgcn_s_wait_asynccnt)
  __builtin_amdgcn_s_wait_asynccnt(N);
#else
  asm volatile("s_wait_asynccnt %0" ::"n"(N) : "memory");
#endif
}

__device__ __forceinline__ unsigned int f32_to_e4m3_sw(float f) {
  unsigned int u = __float_as_uint(f);
  unsigned int sign = (u >> 24) & 0x80u;
  unsigned int au = u & 0x7FFFFFFFu;
  float af = __uint_as_float(au);
  if (af >= 448.0f) return sign | 0x7Eu;
  int e = (int)(au >> 23) - 127;
  if (au == 0u || e <= -11) return sign;
  unsigned int mant = (au & 0x7FFFFFu) | 0x800000u;
  int shift = (e >= -6) ? 20 : (20 + (-6 - e));
  unsigned int lsb    = (mant >> shift) & 1u;
  unsigned int rnd    = (mant >> (shift - 1)) & 1u;
  unsigned int sticky = (mant & ((1u << (shift - 1)) - 1u)) ? 1u : 0u;
  unsigned int q = (mant >> shift) + ((rnd & (lsb | sticky)) ? 1u : 0u);
  if (e >= -6) {
    if (q == 16u) { q = 8u; ++e; }
    int be = e + 7;
    if (be >= 16) return sign | 0x7Eu;
    return sign | (unsigned int)(be << 3) | (q & 7u);
  }
  if (q >= 8u) return sign | 0x08u;
  return sign | q;
}

__device__ __forceinline__ unsigned int pack4_e4m3(float a, float b, float c, float d) {
#if __has_builtin(__builtin_amdgcn_cvt_pk_fp8_f32)
  int r = 0;
  r = __builtin_amdgcn_cvt_pk_fp8_f32(a, b, r, false);
  r = __builtin_amdgcn_cvt_pk_fp8_f32(c, d, r, true);
  return (unsigned int)r;
#else
  return f32_to_e4m3_sw(a)        | (f32_to_e4m3_sw(b) << 8) |
         (f32_to_e4m3_sw(c) << 16) | (f32_to_e4m3_sw(d) << 24);
#endif
}

__global__ __launch_bounds__(256) void hadamard_quant_kernel(
    const float* __restrict__ src, unsigned char* __restrict__ dstq,
    float* __restrict__ dsts, int rows, int K) {
  const int bpr = K >> 6;
  int t = blockIdx.x * blockDim.x + threadIdx.x;
  int row = t / bpr;
  int blk = t - row * bpr;
  if (row >= rows) return;

  float v[64];
  const float4* p4 = (const float4*)(src + (size_t)row * K + (size_t)blk * 64);
#pragma unroll
  for (int i = 0; i < 16; ++i) {
    float4 f = p4[i];
    v[4 * i + 0] = f.x; v[4 * i + 1] = f.y;
    v[4 * i + 2] = f.z; v[4 * i + 3] = f.w;
  }

#pragma unroll
  for (int s = 1; s < 64; s <<= 1) {
#pragma unroll
    for (int i = 0; i < 64; ++i) {
      if ((i & s) == 0) {
        float a = v[i], b = v[i | s];
        v[i] = a + b; v[i | s] = a - b;
      }
    }
  }

  float am = 0.0f;
#pragma unroll
  for (int i = 0; i < 64; ++i) {
    v[i] *= 0.125f;
    am = fmaxf(am, fabsf(v[i]));
  }
#pragma unroll
  for (int m = 1; m < 16; m <<= 1) am = fmaxf(am, __shfl_xor(am, m, 32));

  float scale = fmaxf(am * (1.0f / 448.0f), 1e-12f);
  if (blk == 0) dsts[row] = scale;
  float inv = 1.0f / scale;

  uint4* o4 = (uint4*)(dstq + (size_t)row * K + (size_t)blk * 64);
#pragma unroll
  for (int i = 0; i < 4; ++i) {
    uint4 o;
    o.x = pack4_e4m3(v[16 * i +  0] * inv, v[16 * i +  1] * inv,
                     v[16 * i +  2] * inv, v[16 * i +  3] * inv);
    o.y = pack4_e4m3(v[16 * i +  4] * inv, v[16 * i +  5] * inv,
                     v[16 * i +  6] * inv, v[16 * i +  7] * inv);
    o.z = pack4_e4m3(v[16 * i +  8] * inv, v[16 * i +  9] * inv,
                     v[16 * i + 10] * inv, v[16 * i + 11] * inv);
    o.w = pack4_e4m3(v[16 * i + 12] * inv, v[16 * i + 13] * inv,
                     v[16 * i + 14] * inv, v[16 * i + 15] * inv);
    o4[i] = o;
  }
}

__global__ __launch_bounds__(256) void fp8_gemm_kernel(
    const unsigned char* __restrict__ Aq, const float* __restrict__ As,
    const unsigned char* __restrict__ Bq, const float* __restrict__ Bs,
    const float* __restrict__ bias, float* __restrict__ out,
    int M, int N, int K) {
#if USE_ASYNC_LDS
  __shared__ int ldsA[2][128 * 32];
  __shared__ int ldsB[2][64 * 32];
#else
  __shared__ int ldsA[1][128 * 32];
  __shared__ int ldsB[1][64 * 32];
#endif

  const int t    = threadIdx.x;
  const int wave = t >> 5;
  const int lane = t & 31;
  const int hi   = lane >> 4;
  const int l16  = lane & 15;
  const int m0   = blockIdx.y * 128;
  const int n0   = blockIdx.x * 64;

  const unsigned char* gA = Aq + (size_t)m0 * K;
  const unsigned char* gB = Bq + (size_t)n0 * K;

  v8f acc[4];
  const v8f vzero = {0.f, 0.f, 0.f, 0.f, 0.f, 0.f, 0.f, 0.f};
#pragma unroll
  for (int i = 0; i < 4; ++i) acc[i] = vzero;

  const int arow = wave * 16 + l16;
  const int NT = K >> 7;

  auto compute = [&](const int* lA, const int* lB) {
    union { v16i v; int2 p2[8]; } a;
#pragma unroll
    for (int p = 0; p < 8; ++p) {
      int half = p >> 2, j = p & 3;
      a.p2[p] = *(const int2*)&lA[arow * 32 + (p >> 1) * 8 + hi * 4 + (p & 1) * 2];
    }
#pragma unroll
    for (int nt = 0; nt < 4; ++nt) {
      union { v16i v; int4 p4[4]; } b;
      const int brow = nt * 16 + l16;
#pragma unroll
      for (int g = 0; g < 4; ++g)
        b.p4[g] = *(const int4*)&lB[brow * 32 + g * 8 + hi * 4];
      acc[nt] = __builtin_amdgcn_wmma_f32_16x16x128_fp8_fp8(
          a.v, b.v, (short)0, acc[nt], false, false);
    }
  };

#if USE_ASYNC_LDS
  auto issue_tile = [&](int buf, int k0) {
#pragma unroll
    for (int r = 0; r < 4; ++r) {
      int idx = t + 256 * r, row = idx >> 3, c = idx & 7;
      __builtin_amdgcn_global_load_async_to_lds_b128(
          (GV4*)(gA + (size_t)row * K + k0 + c * 16),
          (LV4*)((char*)&ldsA[buf][0] + idx * 16), 0, 0);
    }
#pragma unroll
    for (int r = 0; r < 2; ++r) {
      int idx = t + 256 * r, row = idx >> 3, c = idx & 7;
      __builtin_amdgcn_global_load_async_to_lds_b128(
          (GV4*)(gB + (size_t)row * K + k0 + c * 16),
          (LV4*)((char*)&ldsB[buf][0] + idx * 16), 0, 0);
    }
  };

  issue_tile(0, 0);
  for (int kt = 0; kt < NT; ++kt) {
    const int cur = kt & 1;
    __syncthreads();
    if (kt + 1 < NT) {
      issue_tile(cur ^ 1, (kt + 1) << 7);
      wait_asynccnt<6>();
    } else {
      wait_asynccnt<0>();
    }
    __syncthreads();
    compute(&ldsA[cur][0], &ldsB[cur][0]);
  }
#else
  uint4 ta[4], tb[2];
  auto load_regs = [&](int k0) {
#pragma unroll
    for (int r = 0; r < 4; ++r) {
      int idx = t + 256 * r, row = idx >> 3, c = idx & 7;
      ta[r] = *(const uint4*)(gA + (size_t)row * K + k0 + c * 16);
    }
#pragma unroll
    for (int r = 0; r < 2; ++r) {
      int idx = t + 256 * r, row = idx >> 3, c = idx & 7;
      tb[r] = *(const uint4*)(gB + (size_t)row * K + k0 + c * 16);
    }
  };
  load_regs(0);
  for (int kt = 0; kt < NT; ++kt) {
    __syncthreads();
#pragma unroll
    for (int r = 0; r < 4; ++r) ((uint4*)&ldsA[0][0])[t + 256 * r] = ta[r];
#pragma unroll
    for (int r = 0; r < 2; ++r) ((uint4*)&ldsB[0][0])[t + 256 * r] = tb[r];
    __syncthreads();
    if (kt + 1 < NT) load_regs((kt + 1) << 7);
    compute(&ldsA[0][0], &ldsB[0][0]);
  }
#endif

  float xs[8];
#pragma unroll
  for (int r = 0; r < 8; ++r) xs[r] = As[m0 + wave * 16 + hi * 8 + r];

#pragma unroll
  for (int nt = 0; nt < 4; ++nt) {
    int n = n0 + nt * 16 + l16;
    float wsn = Bs[n];
    float bn  = bias[n];
#pragma unroll
    for (int r = 0; r < 8; ++r) {
      int m = m0 + wave * 16 + hi * 8 + r;
      out[(size_t)m * N + n] = acc[nt][r] * xs[r] * wsn + bn;
    }
  }
}

extern "C" void kernel_launch(void* const* d_in, const int* in_sizes, int n_in,
                              void* d_out, int out_size, void* d_ws, size_t ws_size,
                              hipStream_t stream) {
  const float* x    = (const float*)d_in[0];
  const float* w    = (const float*)d_in[1];
  const float* bias = (const float*)d_in[2];

  const int N = in_sizes[2];
  const int K = in_sizes[1] / N;
  const int M = in_sizes[0] / K;

  float* out = (float*)d_out;

  size_t off = 0;
  unsigned char* xq = (unsigned char*)d_ws;             off += (size_t)M * K;
  unsigned char* wq = (unsigned char*)d_ws + off;       off += (size_t)N * K;
  off = (off + 255) & ~(size_t)255;
  float* xs = (float*)((char*)d_ws + off);              off += (size_t)M * sizeof(float);
  float* ws = (float*)((char*)d_ws + off);

  const int bpr = K / 64;
  {
    long long tx = (long long)M * bpr;
    hadamard_quant_kernel<<<dim3((unsigned)((tx + 255) / 256)), 256, 0, stream>>>(
        x, xq, xs, M, K);
    long long tw = (long long)N * bpr;
    hadamard_quant_kernel<<<dim3((unsigned)((tw + 255) / 256)), 256, 0, stream>>>(
        w, wq, ws, N, K);
  }

  dim3 grid(N / 64, M / 128);
  fp8_gemm_kernel<<<grid, 256, 0, stream>>>(xq, xs, wq, ws, bias, out, M, N, K);
}
